// WindowAttention3d3_44358422233638
// MI455X (gfx1250) — compile-verified, matrix operand order checked by reading, not run on hardware
//
#include <hip/hip_runtime.h>
#include <hip/hip_bf16.h>

typedef __attribute__((ext_vector_type(16))) _Float16 v16h;
typedef __attribute__((ext_vector_type(8)))  _Float16 v8h;
typedef __attribute__((ext_vector_type(4)))  _Float16 v4h;
typedef __attribute__((ext_vector_type(8)))  float    v8f;
typedef __attribute__((ext_vector_type(4), aligned(4))) float f4u;  // 4B-aligned vec4

#define NTOK 49     // tokens per window (7*7)
#define NP   64     // padded tokens (4 x 16)
#define CDIM 256    // channels
#define HEADS 8
#define HD   32     // head dim
#define NWIN 1024   // windows per image

// workspace layout (bytes)
#define WS_QKVW16  0                             // 768*256 f16  = 393216 B
#define WS_PROJW16 (768 * 256 * 2)               // 256*256 f16  = 131072 B
#define WS_BIASP   (WS_PROJW16 + 256 * 256 * 2)  // 8*64*64 f32  = 131072 B (padded, -1e30)

// ---- WMMA fragment helper --------------------------------------------------
// 16-bit A/B fragment K-layout (CDNA5 ISA 7.12.2): lane<16 holds K in
// {0..7, 16..23}; lane>=16 holds K in {8..15, 24..31}. Caller passes a pointer
// already offset by row*stride + k0 + kbase (kbase = (lane&16)?8:0): the
// fragment is two contiguous 16-byte chunks at p and p+16 (LDS or global).
__device__ __forceinline__ v16h frag16(const _Float16* p) {
  v8h lo = *(const v8h*)(p);
  v8h hi = *(const v8h*)(p + 16);
  return __builtin_shufflevector(lo, hi, 0,1,2,3,4,5,6,7,8,9,10,11,12,13,14,15);
}

__device__ __forceinline__ v8f wmma16(v16h a, v16h b, v8f c) {
  return __builtin_amdgcn_wmma_f32_16x16x32_f16(false, a, false, b, (short)0, c, false, false);
}

// ---- prep kernel: f16 weights + padded relative-position bias ---------------
// biasP[h][64][64]: rpb gather for valid (m,n), -1e30 in padding so the main
// kernel needs NO branches: padding scores are dominated and exp() -> 0.
__global__ void win_attn_prep(const float* __restrict__ qkv_w,
                              const float* __restrict__ proj_w,
                              const float* __restrict__ rpb,
                              const int*   __restrict__ rel,
                              _Float16* __restrict__ qkvw16,
                              _Float16* __restrict__ projw16,
                              float*    __restrict__ biasP)
{
  const int i = blockIdx.x * blockDim.x + threadIdx.x;
  const int NQ = 768 * 256, NPJ = 256 * 256, NB = HEADS * NP * NP;
  if (i < NQ) {
    qkvw16[i] = (_Float16)qkv_w[i];
  } else if (i < NQ + NPJ) {
    const int j = i - NQ;
    projw16[j] = (_Float16)proj_w[j];
  } else if (i < NQ + NPJ + NB) {
    const int j = i - NQ - NPJ;
    const int h = j >> 12, m = (j >> 6) & 63, n = j & 63;
    biasP[j] = (m < NTOK && n < NTOK) ? rpb[rel[m * NTOK + n] * HEADS + h] : -1e30f;
  }
}

// ---- fused window-attention kernel -----------------------------------------
// All GEMMs computed transposed so C-fragment registers (rows) map to the
// contiguous memory axis -> aligned 16B writebacks, no divergent scatters.
// LDS (dynamic, f16, 192 KB):
//   sX   [64][256]  : x tile (phase 1/2), reused as attention output O (3/4)
//   sQ   [64][256]  : scaled Q row-major (B-frags of K*Q^T contiguous)
//   sK   [64][256]  : K row-major (A-frags of K*Q^T contiguous)
//   sVt  [256][64]  : V transposed (A-frags of V^T*P^T contiguous)
//   sP   [8][64][64]: per-head probabilities row-major (B-frags contiguous)
__global__ __launch_bounds__(256, 1)
void win_attn_kernel(const float*    __restrict__ x,
                     const float*    __restrict__ mask,
                     const _Float16* __restrict__ qkvw16,
                     const float*    __restrict__ qkv_b,
                     const _Float16* __restrict__ projw16,
                     const float*    __restrict__ proj_b,
                     const float*    __restrict__ biasP,
                     float* __restrict__ out)
{
  extern __shared__ _Float16 smem[];
  _Float16* sX  = smem;                  // 64*256
  _Float16* sQ  = smem + 1 * NP * CDIM;  // 64*256
  _Float16* sK  = smem + 2 * NP * CDIM;  // 64*256
  _Float16* sVt = smem + 3 * NP * CDIM;  // 256*64
  _Float16* sP  = smem + 4 * NP * CDIM;  // 8*64*64

  const int w    = blockIdx.x;
  const int tid  = threadIdx.x;
  const int wave = __builtin_amdgcn_readfirstlane(tid >> 5);  // SGPR: uniform tiles
  const int lane = tid & 31;
  const int lo16 = lane & 15;
  const int kb   = (lane & 16) ? 8 : 0;   // K-base of this lane's fragment chunks
  const int mho  = (lane & 16) ? 8 : 0;   // C-fragment row offset for upper half

  // ---- Phase 1: x -> LDS as f16 (vectorized), pad rows [49,64) with zeros --
  const float4* xw4 = (const float4*)(x + (size_t)w * NTOK * CDIM);
  for (int i = tid; i < NP * CDIM / 4; i += 256) {
    const int m = i >> 6;                 // 64 float4 per row
    v4h hv = {};
    if (m < NTOK) {
      float4 f = xw4[i];
      hv[0] = (_Float16)f.x; hv[1] = (_Float16)f.y;
      hv[2] = (_Float16)f.z; hv[3] = (_Float16)f.w;
    }
    *(v4h*)(sX + i * 4) = hv;
  }
  __syncthreads();

  // ---- Phase 2: qkv^T = W * x^T   (768x64; 48 channel-tiles x 4 token-tiles)
  // Weight A-fragments loaded once per channel tile, reused over 4 token tiles.
  const float qscale = 0.17677669529663687f;  // hd^-0.5
#pragma unroll 1
  for (int nt = wave * 6; nt < wave * 6 + 6; ++nt) {
    const int n0 = nt * 16;               // SGPR: output-channel tile base
    v16h wf[8];
    const _Float16* wrow = qkvw16 + (size_t)(n0 + lo16) * CDIM + kb;
#pragma unroll
    for (int kk = 0; kk < 8; ++kk) wf[kk] = frag16(wrow + kk * 32);
    const float4 bv0 = *(const float4*)(qkv_b + n0 + mho);
    const float4 bv1 = *(const float4*)(qkv_b + n0 + mho + 4);
    const float bv[8] = {bv0.x, bv0.y, bv0.z, bv0.w, bv1.x, bv1.y, bv1.z, bv1.w};

#pragma unroll 1
    for (int mi = 0; mi < 4; ++mi) {
      v8f c = {};
      const _Float16* brow = sX + (mi * 16 + lo16) * CDIM + kb;
#pragma unroll
      for (int kk = 0; kk < 8; ++kk)
        c = wmma16(wf[kk], frag16(brow + kk * 32), c);

      const int tok = mi * 16 + lo16;     // C column = token (per lane)
      if (n0 < 256) {                     // scalar branch: Q
        v8h pk;
#pragma unroll
        for (int r = 0; r < 8; ++r) pk[r] = (_Float16)((c[r] + bv[r]) * qscale);
        *(v8h*)(sQ + tok * CDIM + n0 + mho) = pk;
      } else if (n0 < 512) {              // scalar branch: K
        v8h pk;
#pragma unroll
        for (int r = 0; r < 8; ++r) pk[r] = (_Float16)(c[r] + bv[r]);
        *(v8h*)(sK + tok * CDIM + (n0 - 256) + mho) = pk;
      } else {                            // scalar branch: V (transposed)
#pragma unroll
        for (int r = 0; r < 8; ++r)
          sVt[(n0 - 512 + mho + r) * NP + tok] = (_Float16)(c[r] + bv[r]);
      }
    }
  }
  __syncthreads();

  // ---- Phase 3: attention, one head per wave; S^T = K * Q^T ----
  const int h  = wave;
  const int mw = w & (NWIN - 1);
  const float* maskw  = mask  + (size_t)mw * NTOK * NTOK;
  const float* biasPH = biasP + (size_t)h * NP * NP;
  _Float16* sPh = sP + h * NP * NP;

  // K and V fragments are head-invariant across query tiles: load once.
  v16h akf[4], avf[4];
#pragma unroll
  for (int ni = 0; ni < 4; ++ni)
    akf[ni] = frag16(sK + (ni * 16 + lo16) * CDIM + h * HD + kb);
#pragma unroll
  for (int dk = 0; dk < 4; ++dk)          // (dj, k-half)
    avf[dk] = frag16(sVt + (h * HD + (dk >> 1) * 16 + lo16) * NP + (dk & 1) * 32 + kb);

#pragma unroll 1
  for (int mi = 0; mi < 4; ++mi) {
    const int m = mi * 16 + lo16;         // query token = C column (per lane)
    v16h bq = frag16(sQ + m * CDIM + h * HD + kb);
    v8f cn[4];
#pragma unroll
    for (int ni = 0; ni < 4; ++ni) {      // key-token tiles = C rows
      v8f z = {};
      cn[ni] = wmma16(akf[ni], bq, z);
    }
    // + padded bias (-1e30 in padding: no branches) + window mask (clamped)
    const float* bP = biasPH + m * NP;                       // 16B-aligned rows
    const float* mC = maskw + (m < NTOK ? m : NTOK - 1) * NTOK;
#pragma unroll
    for (int ni = 0; ni < 4; ++ni) {
      const int nb = ni * 16 + mho;
      float4 b0 = *(const float4*)(bP + nb);
      float4 b1 = *(const float4*)(bP + nb + 4);
      f4u m0 = *(const f4u*)(mC + (nb     < 45 ? nb     : 45));  // clamp: stay in-bounds,
      f4u m1 = *(const f4u*)(mC + (nb + 4 < 45 ? nb + 4 : 45));  // garbage drowned by -1e30
      cn[ni][0] += b0.x + m0[0]; cn[ni][1] += b0.y + m0[1];
      cn[ni][2] += b0.z + m0[2]; cn[ni][3] += b0.w + m0[3];
      cn[ni][4] += b1.x + m1[0]; cn[ni][5] += b1.y + m1[1];
      cn[ni][6] += b1.z + m1[2]; cn[ni][7] += b1.w + m1[3];
    }
    // softmax over n: in-lane reduce + ONE cross-half shuffle each
    float mx = -1e30f;
#pragma unroll
    for (int ni = 0; ni < 4; ++ni)
#pragma unroll
      for (int r = 0; r < 8; ++r) mx = fmaxf(mx, cn[ni][r]);
    mx = fmaxf(mx, __shfl_xor(mx, 16, 32));
    float s = 0.f;
#pragma unroll
    for (int ni = 0; ni < 4; ++ni)
#pragma unroll
      for (int r = 0; r < 8; ++r) {
        float e = __expf(cn[ni][r] - mx);   // padding cols -> 0
        cn[ni][r] = e;
        s += e;
      }
    s += __shfl_xor(s, 16, 32);
    const float inv = 1.0f / s;
#pragma unroll
    for (int ni = 0; ni < 4; ++ni) {
      v8h pk;
#pragma unroll
      for (int r = 0; r < 8; ++r) pk[r] = (_Float16)(cn[ni][r] * inv);
      *(v8h*)(sPh + m * NP + ni * 16 + mho) = pk;   // row-major P[m][n]
    }
    // O^T = V^T * P^T : A preloaded (avf), B from row-major sP (contiguous)
    v16h bp0 = frag16(sPh + m * NP + kb);
    v16h bp1 = frag16(sPh + m * NP + 32 + kb);
#pragma unroll
    for (int dj = 0; dj < 2; ++dj) {
      v8f o = {};
      o = wmma16(avf[dj * 2 + 0], bp0, o);
      o = wmma16(avf[dj * 2 + 1], bp1, o);
      v8h pk;
#pragma unroll
      for (int r = 0; r < 8; ++r) pk[r] = (_Float16)o[r];
      *(v8h*)(sX + m * CDIM + h * HD + dj * 16 + mho) = pk;  // O row-major
    }
  }
  __syncthreads();

  // ---- Phase 4: out^T = Wproj * O^T + b  (16 channel-tiles x 4 token-tiles)
  float* outw = out + (size_t)w * NTOK * CDIM;
#pragma unroll 1
  for (int nt = wave * 2; nt < wave * 2 + 2; ++nt) {
    const int n0 = nt * 16;
    v16h wf[8];
    const _Float16* wrow = projw16 + (size_t)(n0 + lo16) * CDIM + kb;
#pragma unroll
    for (int kk = 0; kk < 8; ++kk) wf[kk] = frag16(wrow + kk * 32);
    const float4 bv0 = *(const float4*)(proj_b + n0 + mho);
    const float4 bv1 = *(const float4*)(proj_b + n0 + mho + 4);

#pragma unroll 1
    for (int mi = 0; mi < 4; ++mi) {
      v8f c = {};
      const _Float16* brow = sX + (mi * 16 + lo16) * CDIM + kb;
#pragma unroll
      for (int kk = 0; kk < 8; ++kk)
        c = wmma16(wf[kk], frag16(brow + kk * 32), c);

      const int tok = mi * 16 + lo16;
      if (tok < NTOK) {                   // two aligned b128 stores per lane
        float4 o0 = make_float4(c[0] + bv0.x, c[1] + bv0.y, c[2] + bv0.z, c[3] + bv0.w);
        float4 o1 = make_float4(c[4] + bv1.x, c[5] + bv1.y, c[6] + bv1.z, c[7] + bv1.w);
        float* dst = outw + (size_t)tok * CDIM + n0 + mho;
        *(float4*)(dst)     = o0;
        *(float4*)(dst + 4) = o1;
      }
    }
  }
}

extern "C" void kernel_launch(void* const* d_in, const int* in_sizes, int n_in,
                              void* d_out, int out_size, void* d_ws, size_t ws_size,
                              hipStream_t stream) {
  const float* x      = (const float*)d_in[0];
  const float* mask   = (const float*)d_in[1];
  const float* qkv_w  = (const float*)d_in[2];
  const float* qkv_b  = (const float*)d_in[3];
  const float* proj_w = (const float*)d_in[4];
  const float* proj_b = (const float*)d_in[5];
  const float* rpb    = (const float*)d_in[6];
  const int*   rel    = (const int*)d_in[7];
  float* out = (float*)d_out;

  char* ws = (char*)d_ws;
  _Float16* qkvw16  = (_Float16*)(ws + WS_QKVW16);
  _Float16* projw16 = (_Float16*)(ws + WS_PROJW16);
  float*    biasP   = (float*)(ws + WS_BIASP);

  const int prep_elems = 768 * 256 + 256 * 256 + HEADS * NP * NP;
  win_attn_prep<<<(prep_elems + 255) / 256, 256, 0, stream>>>(
      qkv_w, proj_w, rpb, rel, qkvw16, projw16, biasP);

  const int nwin = in_sizes[0] / (NTOK * CDIM);  // 4096 windows
  const size_t shmem = (size_t)(4 * NP * CDIM + HEADS * NP * NP) * sizeof(_Float16); // 192 KB

  hipFuncSetAttribute(reinterpret_cast<const void*>(win_attn_kernel),
                      hipFuncAttributeMaxDynamicSharedMemorySize, (int)shmem);
  win_attn_kernel<<<nwin, 256, shmem, stream>>>(x, mask, qkvw16, qkv_b,
                                                projw16, proj_b, biasP, out);
}
